// Cross_Attention_Fourier_73650099191989
// MI455X (gfx1250) — matrix operand order repaired and checked by reading, not compiled, not run on hardware
//
#include <hip/hip_runtime.h>
#include <hip/hip_bf16.h>

typedef __bf16 bf16;
typedef __bf16 v16bf __attribute__((ext_vector_type(16)));
typedef __bf16 v8bf  __attribute__((ext_vector_type(8)));
typedef float  v8f   __attribute__((ext_vector_type(8)));
typedef unsigned int u32x4 __attribute__((ext_vector_type(4)));
typedef int          i32x8 __attribute__((ext_vector_type(8)));
typedef int          i32x4 __attribute__((ext_vector_type(4)));

#define NDIM 512
#define NSEQ 1024
#define BATCH 4
#define HEADS 8
#define DH 64

#define HAVE_TDM __has_builtin(__builtin_amdgcn_tensor_load_to_lds)

__device__ __forceinline__ bf16 f2bf(float x) { return (bf16)x; }

// ---------------- WMMA fragment loaders (gfx1250 wave32 layouts) ----------------
// A operand, 16x32 bf16 tile, row-major source (lda elements). lane&15 = M row.
// k = 16*(lane>>4) + e  -> two contiguous 16B loads.
__device__ __forceinline__ v16bf load_A16x32(const bf16* A, int lda, int k0, int lane) {
    int half = lane >> 4;
    const bf16* p = A + (size_t)(lane & 15) * lda + k0 + 16 * half;
    v8bf lo = *(const v8bf*)(p);
    v8bf hi = *(const v8bf*)(p + 8);
    return __builtin_shufflevector(lo, hi, 0,1,2,3,4,5,6,7,8,9,10,11,12,13,14,15);
}
// B operand, 32x16 bf16 tile. Source is "Bt": [N][K] row-major (ldb elements).
// lane&15 = N col, k = k0 + 16*(lane>>4) + e  -> one contiguous 32B load.
__device__ __forceinline__ v16bf load_B32x16(const bf16* Bt, int ldb, int k0, int lane) {
    const bf16* p = Bt + (size_t)(lane & 15) * ldb + k0 + 16 * (lane >> 4);
    return *(const v16bf*)(p);
}
__device__ __forceinline__ v8f wmma_bf16(v16bf a, v16bf b, v8f c) {
    return __builtin_amdgcn_wmma_f32_16x16x32_bf16(false, a, false, b, (short)0, c, false, false);
}

#if HAVE_TDM
// ---- Tensor Data Mover: 2-D tile load (rows x 32K bf16) into LDS ----------------
// D# packing per CDNA5 ISA (normal mode, <=2D,
// groups 2/3 and extra group zero). data_size=1 (2 bytes). Caller wave-uniform.
__device__ __forceinline__ void tdm_load_2d(unsigned lds_addr, const bf16* gptr,
                                            int tensor_d0, int tensor_d1,
                                            int tile_d0, int tile_d1, int stride_d0) {
    unsigned long long ga = (unsigned long long)(uintptr_t)gptr;
    u32x4 g0;
    g0.x = 1u;                                                 // count=1, user mode
    g0.y = lds_addr;                                           // lds_addr [63:32]
    g0.z = (unsigned)(ga & 0xFFFFFFFFu);                       // global_addr [95:64]
    g0.w = (unsigned)((ga >> 32) & 0x01FFFFFFu) | (2u << 30);  // addr[56:32] | type=2
    i32x8 g1;
    g1[0] = (1 << 16);                                         // data_size=2B, mask=0
    g1[1] = (tensor_d0 & 0xFFFF) << 16;                        // tensor_dim0 lo16
    g1[2] = ((tensor_d0 >> 16) & 0xFFFF) | ((tensor_d1 & 0xFFFF) << 16);
    g1[3] = ((tensor_d1 >> 16) & 0xFFFF) | ((tile_d0 & 0xFFFF) << 16);
    g1[4] = (tile_d1 & 0xFFFF);                                // tile_dim1, tile_dim2=0
    g1[5] = stride_d0;                                         // dim0 stride lo32
    g1[6] = 0;
    g1[7] = 0;
    i32x4 gz4 = {0, 0, 0, 0};
    i32x8 gz8 = {0, 0, 0, 0, 0, 0, 0, 0};
    __builtin_amdgcn_tensor_load_to_lds(g0, g1, gz4, gz4, gz8, 0);
}
__device__ __forceinline__ unsigned lds_off(const void* p) {
    // LDS aperture carries the workgroup-relative offset in the low 32 bits
    return (unsigned)(uintptr_t)p;
}
#endif

// ---------------- weight convert + transpose: wt[n][k] = bf16(w[k][n]) (512x512) ----
__global__ __launch_bounds__(256) void transpose_to_bf16(const float* __restrict__ w,
                                                         bf16* __restrict__ wt) {
    int i = blockIdx.x * 256 + threadIdx.x;       // 0..262143
    int n = i >> 9, k = i & 511;
    wt[i] = f2bf(w[k * NDIM + n]);
}

// ---------------- LayerNorm (512 cols, ddof=0) -> bf16, one wave per row ----------
__global__ __launch_bounds__(256) void ln_to_bf16(const float* __restrict__ x,
                                                  const float* __restrict__ g,
                                                  const float* __restrict__ bta,
                                                  bf16* __restrict__ y) {
    int wave = threadIdx.x >> 5, lane = threadIdx.x & 31;
    int row = blockIdx.x * 8 + wave;
    const float* xr = x + (size_t)row * NDIM;
    float v[16], s = 0.f, sq = 0.f;
#pragma unroll
    for (int i = 0; i < 16; ++i) { float t = xr[lane + 32 * i]; v[i] = t; s += t; sq += t * t; }
#pragma unroll
    for (int m = 1; m < 32; m <<= 1) { s += __shfl_xor(s, m, 32); sq += __shfl_xor(sq, m, 32); }
    float mean = s * (1.f / NDIM);
    float var  = sq * (1.f / NDIM) - mean * mean;
    float rstd = rsqrtf(var + 1e-5f);
#pragma unroll
    for (int i = 0; i < 16; ++i) {
        int c = lane + 32 * i;
        y[(size_t)row * NDIM + c] = f2bf((v[i] - mean) * rstd * g[c] + bta[c]);
    }
}

// ---------------- bf16 GEMM: C[M,N] = A[M,K] @ Bt[N,K]^T (+bias) ------------------
// Block = 64x128 C tile, 8 waves; wave (mt = w&3, nh = w>>2) computes 16x64
// (4 accumulators reusing one A fragment). K streamed in 32-wide chunks,
// double-buffered in LDS by the Tensor Data Mover (TENSORcnt-tracked).
// epilogue: 0 = f32 store, 1 = bf16 store, 2 = exact-GELU -> bf16,
//           3 = V permute store: VpT[b][h][d][(N-n)%N] (head transpose + reversal)
__global__ __launch_bounds__(256) void gemm_bf16(const bf16* __restrict__ A,
                                                 const bf16* __restrict__ Bt,
                                                 const float* __restrict__ bias,
                                                 float* __restrict__ Cf,
                                                 bf16* __restrict__ Cb,
                                                 int N, int Kd, int epilogue) {
    int lane = threadIdx.x & 31;
    int wave = threadIdx.x >> 5;
    int mt = wave & 3;        // 4 row tiles of 16
    int nh = wave >> 2;       // 2 col halves of 64
    int bm0 = blockIdx.x * 64;
    int bn0 = blockIdx.y * 128;
    v8f acc[4] = {};

#if HAVE_TDM
    __shared__ __align__(32) bf16 Abuf[2][64][32];    //  8 KB
    __shared__ __align__(32) bf16 Bbuf[2][128][32];   // 16 KB
    int nk = Kd >> 5;
    if (wave == 0) {   // prologue: stage chunk 0
        tdm_load_2d(lds_off(&Abuf[0][0][0]), A  + (size_t)bm0 * Kd, Kd, 4096, 32, 64,  Kd);
        tdm_load_2d(lds_off(&Bbuf[0][0][0]), Bt + (size_t)bn0 * Kd, Kd, N * 8, 32, 128, Kd);
    }
    for (int kc = 0; kc < nk; ++kc) {
        int cur = kc & 1, nxt = cur ^ 1;
        __syncthreads();                       // everyone done reading buffer `nxt`
        if (wave == 0) {
            if (kc + 1 < nk) {
                const bf16* gA = A  + (size_t)bm0 * Kd + (kc + 1) * 32;
                const bf16* gB = Bt + (size_t)bn0 * Kd + (kc + 1) * 32;
                tdm_load_2d(lds_off(&Abuf[nxt][0][0]), gA, Kd, 4096, 32, 64,  Kd);
                tdm_load_2d(lds_off(&Bbuf[nxt][0][0]), gB, Kd, N * 8, 32, 128, Kd);
                __builtin_amdgcn_s_wait_tensorcnt(2);   // chunk kc landed, kc+1 in flight
            } else {
                __builtin_amdgcn_s_wait_tensorcnt(0);
            }
        }
        __syncthreads();                       // publish chunk kc
        v16bf af = load_A16x32(&Abuf[cur][mt * 16][0], 32, 0, lane);
#pragma unroll
        for (int t = 0; t < 4; ++t) {
            v16bf bfr = load_B32x16(&Bbuf[cur][nh * 64 + t * 16][0], 32, 0, lane);
            acc[t] = wmma_bf16(af, bfr, acc[t]);
        }
    }
#else
    const bf16* Arow = A + (size_t)(bm0 + mt * 16) * Kd;
    for (int k0 = 0; k0 < Kd; k0 += 32) {
        v16bf af = load_A16x32(Arow, Kd, k0, lane);
#pragma unroll
        for (int t = 0; t < 4; ++t) {
            const bf16* Brow = Bt + (size_t)(bn0 + nh * 64 + t * 16) * Kd;
            v16bf bfr = load_B32x16(Brow, Kd, k0, lane);
            acc[t] = wmma_bf16(af, bfr, acc[t]);
        }
    }
#endif

    int half = lane >> 4;
#pragma unroll
    for (int t = 0; t < 4; ++t) {
        int col = bn0 + nh * 64 + t * 16 + (lane & 15);
        float bv = bias ? bias[col] : 0.0f;
#pragma unroll
        for (int v = 0; v < 8; ++v) {
            int row = bm0 + mt * 16 + v + 8 * half;
            float x = acc[t][v] + bv;
            if (epilogue == 0) {
                Cf[(size_t)row * N + col] = x;
            } else if (epilogue == 1) {
                Cb[(size_t)row * N + col] = f2bf(x);
            } else if (epilogue == 2) {
                float gl = 0.5f * x * (1.0f + erff(x * 0.70710678118654752f));
                Cb[(size_t)row * N + col] = f2bf(gl);
            } else {  // epilogue == 3
                int br = row >> 10, nr = row & (NSEQ - 1);
                int hh = col >> 6,  dd = col & (DH - 1);
                int mrev = (NSEQ - nr) & (NSEQ - 1);
                Cb[((size_t)((br * HEADS + hh) * DH + dd)) * NSEQ + mrev] = f2bf(x);
            }
        }
    }
}

// ---------------- fused Fourier attention:  out = softmax(|Q K^T|)/8 @ Vp ---------
// grid (NSEQ/16, HEADS, BATCH), 256 threads (8 waves). Each block: 16 query rows.
__global__ __launch_bounds__(256) void attn_fourier(const bf16* __restrict__ Q,
                                                    const bf16* __restrict__ Kb,
                                                    const bf16* __restrict__ VpT,
                                                    float* __restrict__ out) {
    __shared__ __align__(32) bf16 attnS[16 * NSEQ];   // 32KB probability tile
    __shared__ float smax[8][16];
    __shared__ float ssum[8][16];
    __shared__ float obuf[4][8][32];
    int lane = threadIdx.x & 31;
    int wave = threadIdx.x >> 5;
    int half = lane >> 4;
    int n0 = blockIdx.x * 16;
    int h  = blockIdx.y;
    int b  = blockIdx.z;
    const bf16* Qbase = Q  + ((size_t)(b * NSEQ + n0)) * NDIM + h * DH;  // lda = NDIM
    const bf16* Kbase = Kb + ((size_t)(b * NSEQ)) * NDIM + h * DH;       // Bt view, ldb = NDIM

    // --- phase 1: S row-block (16 x 1024); wave w covers cols [w*128, w*128+128)
    v8f acc[8] = {};
#pragma unroll
    for (int kk = 0; kk < DH; kk += 32) {
        v16bf af = load_A16x32(Qbase, NDIM, kk, lane);
#pragma unroll
        for (int t = 0; t < 8; ++t) {
            const bf16* Bp = Kbase + (size_t)(wave * 128 + t * 16) * NDIM;
            v16bf bfr = load_B32x16(Bp, NDIM, kk, lane);
            acc[t] = wmma_bf16(af, bfr, acc[t]);
        }
    }
    // --- softmax over |S| (full 1024-wide rows, 8 waves cooperate)
    float rmax[8];
#pragma unroll
    for (int v = 0; v < 8; ++v) {
        float m = -1e30f;
#pragma unroll
        for (int t = 0; t < 8; ++t) m = fmaxf(m, fabsf(acc[t][v]));
#pragma unroll
        for (int sh = 1; sh < 16; sh <<= 1) m = fmaxf(m, __shfl_xor(m, sh, 16));
        rmax[v] = m;
    }
    if ((lane & 15) == 0)
        for (int v = 0; v < 8; ++v) smax[wave][8 * half + v] = rmax[v];
    __syncthreads();
    float gmax[8];
#pragma unroll
    for (int v = 0; v < 8; ++v) {
        float m = smax[0][8 * half + v];
#pragma unroll
        for (int w = 1; w < 8; ++w) m = fmaxf(m, smax[w][8 * half + v]);
        gmax[v] = m;
    }
    float rsum[8];
#pragma unroll
    for (int v = 0; v < 8; ++v) rsum[v] = 0.f;
#pragma unroll
    for (int t = 0; t < 8; ++t)
#pragma unroll
        for (int v = 0; v < 8; ++v) {
            float e = __expf(fabsf(acc[t][v]) - gmax[v]);
            acc[t][v] = e;
            rsum[v] += e;
        }
#pragma unroll
    for (int v = 0; v < 8; ++v) {
        float s = rsum[v];
#pragma unroll
        for (int sh = 1; sh < 16; sh <<= 1) s += __shfl_xor(s, sh, 16);
        rsum[v] = s;
    }
    if ((lane & 15) == 0)
        for (int v = 0; v < 8; ++v) ssum[wave][8 * half + v] = rsum[v];
    __syncthreads();
    float gscale[8];
#pragma unroll
    for (int v = 0; v < 8; ++v) {
        float s = 0.f;
        for (int w = 0; w < 8; ++w) s += ssum[w][8 * half + v];
        gscale[v] = 0.125f / s;   // softmax denom and 1/sqrt(64)
    }
    // stage attn tile to LDS as WMMA-A source (row-major 16x1024 bf16)
#pragma unroll
    for (int t = 0; t < 8; ++t)
#pragma unroll
        for (int v = 0; v < 8; ++v) {
            int row = v + 8 * half;
            int col = wave * 128 + t * 16 + (lane & 15);
            attnS[row * NSEQ + col] = f2bf(acc[t][v] * gscale[v]);
        }
    __syncthreads();

    // --- phase 2: out(16x64) = attnS(16x1024) @ Vp(1024x64); VpT is [d][m]
    int ntile = wave & 3;   // d tile
    int mh    = wave >> 2;  // m half
    const bf16* Vbase = VpT + (size_t)((b * HEADS + h) * DH + ntile * 16) * NSEQ;
    v8f oacc = {};
    for (int k0 = mh * 512; k0 < mh * 512 + 512; k0 += 32) {
        v16bf af = load_A16x32(attnS, NSEQ, k0, lane);
        v16bf bfr = load_B32x16(Vbase, NSEQ, k0, lane);
        oacc = wmma_bf16(af, bfr, oacc);
    }
    if (mh == 1)
#pragma unroll
        for (int v = 0; v < 8; ++v) obuf[ntile][v][lane] = oacc[v];
    __syncthreads();
    if (mh == 0) {
#pragma unroll
        for (int v = 0; v < 8; ++v) {
            float o = oacc[v] + obuf[ntile][v][lane];
            int row = n0 + v + 8 * half;
            int col = h * DH + ntile * 16 + (lane & 15);
            out[(size_t)(b * NSEQ + row) * NDIM + col] = o;
        }
    }
}

// ---------------- global per-sample mean/sumsq reduction --------------------------
__global__ void stats_init(float* stats) { if (threadIdx.x < 8) stats[threadIdx.x] = 0.f; }

__global__ __launch_bounds__(256) void stats_reduce(const float* __restrict__ x,
                                                    float* __restrict__ stats) {
    int b = blockIdx.y;
    const float* xb = x + (size_t)b * NSEQ * NDIM;
    size_t base = (size_t)blockIdx.x * 4096 + threadIdx.x;
    float s = 0.f, sq = 0.f;
#pragma unroll
    for (int i = 0; i < 16; ++i) { float t = xb[base + i * 256]; s += t; sq += t * t; }
#pragma unroll
    for (int m = 1; m < 32; m <<= 1) { s += __shfl_xor(s, m, 32); sq += __shfl_xor(sq, m, 32); }
    __shared__ float ls[8], lq[8];
    int lane = threadIdx.x & 31, wave = threadIdx.x >> 5;
    if (lane == 0) { ls[wave] = s; lq[wave] = sq; }
    __syncthreads();
    if (threadIdx.x == 0) {
        float S = 0.f, Qq = 0.f;
        for (int w = 0; w < 8; ++w) { S += ls[w]; Qq += lq[w]; }
        atomicAdd(&stats[b * 2], S);
        atomicAdd(&stats[b * 2 + 1], Qq);
    }
}

// ---------------- FiLM time-embedding MLP (tiny, f32 VALU) ------------------------
__global__ __launch_bounds__(256) void film1_kernel(const float* __restrict__ temb,
                                                    const float* __restrict__ w1,
                                                    const float* __restrict__ b1,
                                                    float* __restrict__ t1) {
    int b = blockIdx.y;
    int j = blockIdx.x * 256 + threadIdx.x;   // 0..1023
    const float* te = temb + b * 512;
    float s = b1[j];
    for (int k = 0; k < 512; ++k) s += te[k] * w1[k * 1024 + j];
    t1[b * 1024 + j] = s / (1.f + __expf(-s));   // SiLU
}
__global__ __launch_bounds__(256) void film2_kernel(const float* __restrict__ t1,
                                                    const float* __restrict__ w2,
                                                    const float* __restrict__ b2,
                                                    float* __restrict__ t2) {
    int b = blockIdx.y;
    int j = blockIdx.x * 256 + threadIdx.x;
    const float* tb = t1 + b * 1024;
    float s = b2[j];
    for (int k = 0; k < 1024; ++k) s += tb[k] * w2[k * 1024 + j];
    t2[b * 1024 + j] = s;
}

// ---------------- (x - mu)/sd(ddof=1) * std_t + mean_t -> bf16 --------------------
__global__ __launch_bounds__(256) void norm_film(const float* __restrict__ x,
                                                 const float* __restrict__ stats,
                                                 const float* __restrict__ t2,
                                                 bf16* __restrict__ y) {
    size_t i = (size_t)blockIdx.x * 256 + threadIdx.x;
    int b = (int)(i >> 19);          // / (1024*512)
    int c = (int)(i & (NDIM - 1));
    float S = stats[b * 2], Qq = stats[b * 2 + 1];
    const float Nf = (float)(NSEQ * NDIM);
    float mu  = S / Nf;
    float var = (Qq - S * mu) / (Nf - 1.f);
    float rsd = rsqrtf(var);
    float mean_t = t2[b * 1024 + c];
    float std_t  = t2[b * 1024 + NDIM + c];
    y[i] = f2bf((x[i] - mu) * rsd * std_t + mean_t);
}

// =================================================================================
extern "C" void kernel_launch(void* const* d_in, const int* in_sizes, int n_in,
                              void* d_out, int out_size, void* d_ws, size_t ws_size,
                              hipStream_t stream) {
    (void)in_sizes; (void)n_in; (void)out_size; (void)ws_size;
    const float* con      = (const float*)d_in[0];
    const float* diff     = (const float*)d_in[1];
    const float* temb     = (const float*)d_in[2];
    const float* ln_con_g = (const float*)d_in[3];
    const float* ln_con_b = (const float*)d_in[4];
    const float* ln_dif_g = (const float*)d_in[5];
    const float* ln_dif_b = (const float*)d_in[6];
    const float* wq       = (const float*)d_in[7];
    const float* wk       = (const float*)d_in[8];
    const float* wv       = (const float*)d_in[9];
    const float* w_out    = (const float*)d_in[10];
    const float* b_out    = (const float*)d_in[11];
    const float* w_emd1   = (const float*)d_in[12];
    const float* b_emd1   = (const float*)d_in[13];
    const float* w_emd2   = (const float*)d_in[14];
    const float* b_emd2   = (const float*)d_in[15];
    const float* mlp_ln_g = (const float*)d_in[16];
    const float* mlp_ln_b = (const float*)d_in[17];
    const float* mlp_w1   = (const float*)d_in[18];
    const float* mlp_b1   = (const float*)d_in[19];
    const float* mlp_w2   = (const float*)d_in[20];
    const float* mlp_b2   = (const float*)d_in[21];

    char* ws = (char*)d_ws;
    size_t off = 0;
    auto alloc = [&](size_t bytes) -> char* {
        char* p = ws + off;
        off = (off + bytes + 255) & ~(size_t)255;
        return p;
    };
    const size_t WSZ = (size_t)NDIM * NDIM * sizeof(bf16);      // 512 KB
    const size_t ASZ = (size_t)BATCH * NSEQ * NDIM;             // 2M elems
    bf16* wqT  = (bf16*)alloc(WSZ);
    bf16* wkT  = (bf16*)alloc(WSZ);
    bf16* wvT  = (bf16*)alloc(WSZ);
    bf16* woT  = (bf16*)alloc(WSZ);
    bf16* w1T  = (bf16*)alloc(WSZ);
    bf16* w2T  = (bf16*)alloc(WSZ);
    bf16* QLN  = (bf16*)alloc(ASZ * 2);
    bf16* KVLN = (bf16*)alloc(ASZ * 2);
    bf16* Qb   = (bf16*)alloc(ASZ * 2);
    bf16* Kb   = (bf16*)alloc(ASZ * 2);
    bf16* VpT  = (bf16*)alloc(ASZ * 2);
    bf16* X1   = (bf16*)alloc(ASZ * 2);
    bf16* Hb   = (bf16*)alloc(ASZ * 2);
    bf16* Gb   = (bf16*)alloc(ASZ * 2);
    float* attnout = (float*)alloc(ASZ * 4);
    float* OUT1    = (float*)alloc(ASZ * 4);
    float* stats   = (float*)alloc(8 * 4);
    float* t1      = (float*)alloc(BATCH * 1024 * 4);
    float* t2      = (float*)alloc(BATCH * 1024 * 4);

    // weights -> bf16, transposed to [N][K] for contiguous WMMA-B / TDM tiles
    transpose_to_bf16<<<1024, 256, 0, stream>>>(wq, wqT);
    transpose_to_bf16<<<1024, 256, 0, stream>>>(wk, wkT);
    transpose_to_bf16<<<1024, 256, 0, stream>>>(wv, wvT);
    transpose_to_bf16<<<1024, 256, 0, stream>>>(w_out, woT);
    transpose_to_bf16<<<1024, 256, 0, stream>>>(mlp_w1, w1T);
    transpose_to_bf16<<<1024, 256, 0, stream>>>(mlp_w2, w2T);

    // LayerNorms -> bf16 activations
    ln_to_bf16<<<512, 256, 0, stream>>>(diff, ln_dif_g, ln_dif_b, QLN);
    ln_to_bf16<<<512, 256, 0, stream>>>(con,  ln_con_g, ln_con_b, KVLN);

    // QKV projections (V stored head-transposed + row-reversed: Vp[m] = v[(N-m)%N])
    dim3 gg(64, 4);   // M/64 x N/128
    gemm_bf16<<<gg, 256, 0, stream>>>(QLN,  wqT, nullptr, nullptr, Qb,  NDIM, NDIM, 1);
    gemm_bf16<<<gg, 256, 0, stream>>>(KVLN, wkT, nullptr, nullptr, Kb,  NDIM, NDIM, 1);
    gemm_bf16<<<gg, 256, 0, stream>>>(KVLN, wvT, nullptr, nullptr, VpT, NDIM, NDIM, 3);

    // fused attention (FFT identity: softmax(|QK^T|)/8 @ Vp)
    dim3 ga(NSEQ / 16, HEADS, BATCH);
    attn_fourier<<<ga, 256, 0, stream>>>(Qb, Kb, VpT, attnout);

    // per-sample standardization stats (ddof=1)
    stats_init<<<1, 8, 0, stream>>>(stats);
    dim3 gs(128, BATCH);
    stats_reduce<<<gs, 256, 0, stream>>>(attnout, stats);

    // FiLM from time embedding
    dim3 gf(4, BATCH);
    film1_kernel<<<gf, 256, 0, stream>>>(temb, w_emd1, b_emd1, t1);
    film2_kernel<<<gf, 256, 0, stream>>>(t1, w_emd2, b_emd2, t2);

    // standardize + FiLM -> bf16
    norm_film<<<8192, 256, 0, stream>>>(attnout, stats, t2, X1);

    // output projection, MLP (LN -> GEMM+GELU -> GEMM)
    gemm_bf16<<<gg, 256, 0, stream>>>(X1, woT, b_out, OUT1, nullptr, NDIM, NDIM, 0);
    ln_to_bf16<<<512, 256, 0, stream>>>(OUT1, mlp_ln_g, mlp_ln_b, Hb);
    gemm_bf16<<<gg, 256, 0, stream>>>(Hb, w1T, mlp_b1, nullptr, Gb, NDIM, NDIM, 2);
    gemm_bf16<<<gg, 256, 0, stream>>>(Gb, w2T, mlp_b2, (float*)d_out, nullptr, NDIM, NDIM, 0);
}
